// FrameTransformer_25434796327661
// MI455X (gfx1250) — matrix operand order repaired and checked by reading, not compiled, not run on hardware
//
#include <hip/hip_runtime.h>
#include <hip/hip_bf16.h>
#include <math.h>

// Model dims
#define Bb 2
#define Cc 2
#define Ff 1024
#define Ww 1024
#define Hh 4
#define Ee 4
#define Ll 15
#define OCc 2
#define NBb 3
#define Dd (Ff / Hh)   // 256

typedef __attribute__((ext_vector_type(16))) __bf16 v16bf;
typedef __attribute__((ext_vector_type(8)))  __bf16 bf16x8;
typedef __attribute__((ext_vector_type(4)))  __bf16 bf16x4;
typedef __attribute__((ext_vector_type(2)))  __bf16 bf16x2;
typedef __attribute__((ext_vector_type(8)))  float  v8f;

#if defined(__gfx1250__) && __has_builtin(__builtin_amdgcn_global_load_async_to_lds_b128)
#define HAVE_ASYNC_LDS 1
typedef int v4i_ __attribute__((vector_size(16)));
typedef __attribute__((address_space(1))) v4i_ g_v4i;   // global 16B chunk
typedef __attribute__((address_space(3))) v4i_ l_v4i;   // LDS 16B chunk
#if __has_builtin(__builtin_amdgcn_s_wait_asynccnt)
#define WAIT_ASYNC(n) __builtin_amdgcn_s_wait_asynccnt(n)
#else
#define WAIT_ASYNC(n) asm volatile("s_wait_asynccnt %0" :: "i"(n) : "memory")
#endif
#else
#define HAVE_ASYNC_LDS 0
#define WAIT_ASYNC(n) ((void)0)
#endif

__device__ __forceinline__ __bf16 f2bf(float f) {
  unsigned u = __builtin_bit_cast(unsigned, f);
  u += 0x7FFFu + ((u >> 16) & 1u);           // round-to-nearest-even
  unsigned short s = (unsigned short)(u >> 16);
  return __builtin_bit_cast(__bf16, s);
}

__device__ __forceinline__ float gelu_exact(float x) {
  return 0.5f * x * (1.0f + erff(x * 0.70710678118654752f));
}

// ---------------------------------------------------------------------------
// WMMA bf16 GEMM.  All operands bf16:
//   A: M x K row-major (lda).   B: N x K row-major (ldb)  [i.e. "B transposed"].
// Batch z: off(op) = (z / zdiv)*s1 + (z % zdiv)*s2 ; bias off = (z % zdivA)*sBias.
// OUTMODE 0: C f32, M x N row-major (ldc).
// OUTMODE 1: C bf16, transposed: addr = n*ldc + m (packed bf16x8 stores).
// Tile 128(M) x 256(N) x 32(K); 8 waves (wave32) as 2(M) x 4(N);
// wave tile 64x64 = 4x4 fragments of 16x16 -> 16 WMMA per wave per K-step.
// Staging: global->LDS via GLOBAL_LOAD_ASYNC_TO_LDS_B128, double-buffered,
// steady-state loop peeled (branch-free): stage k+1, s_wait_asynccnt 6, barrier.
// ---------------------------------------------------------------------------
#define BM 128
#define BN 256
#define BKK 32
#define LPAD 8

template<bool FUSE_GELU, bool HAS_BIAS, int OUTMODE>
__global__ __launch_bounds__(256) void gemm_bf16_wmma(
    const __bf16* __restrict__ Ag, const __bf16* __restrict__ Bg,
    const float* __restrict__ biasg, void* __restrict__ Cg,
    int M, int N, int K, int lda, int ldb, int ldc,
    int zdivA, long long sA1, long long sA2,
    int zdivB, long long sB1, long long sB2,
    int zdivC, long long sC1, long long sC2,
    long long sBias)
{
  (void)M; (void)N;
  __shared__ __bf16 sA[2][BM][BKK + LPAD];
  __shared__ __bf16 sB[2][BN][BKK + LPAD];

  const int z = blockIdx.z;
  const __bf16* A = Ag + (long long)(z / zdivA) * sA1 + (long long)(z % zdivA) * sA2;
  const __bf16* B = Bg + (long long)(z / zdivB) * sB1 + (long long)(z % zdivB) * sB2;
  const float* bias = biasg ? (biasg + (long long)(z % zdivA) * sBias) : nullptr;
  const long long coff = (long long)(z / zdivC) * sC1 + (long long)(z % zdivC) * sC2;

  const int m0 = blockIdx.y * BM;
  const int n0 = blockIdx.x * BN;
  const int tid  = threadIdx.x;
  const int lane = tid & 31;
  const int wv   = tid >> 5;
  const int mw   = (wv >> 2) * 64;   // wave M offset in tile (0/64)
  const int nw   = (wv & 3) * 64;    // wave N offset in tile (0..192)

  const __bf16* Ablk = A + (long long)m0 * lda;
  const __bf16* Bblk = B + (long long)n0 * ldb;

  // staging: 16B chunks; A tile 128x32 (2 per thread), B tile 256x32 (4 per thread)
  const int srow = tid >> 2;            // 0..63
  const int sko  = (tid & 3) * 8;       // 0,8,16,24
  auto stageA = [&](int kt, int p) {
    const long long kofs = (long long)kt * BKK + sko;
    #pragma unroll
    for (int it = 0; it < 2; it++) {
      const int row = srow + it * 64;
      const __bf16* src = Ablk + (long long)row * lda + kofs;
      __bf16* dst = &sA[p][row][sko];
#if HAVE_ASYNC_LDS
      __builtin_amdgcn_global_load_async_to_lds_b128((g_v4i*)src, (l_v4i*)dst, 0, 0);
#else
      *(bf16x8*)dst = *(const bf16x8*)src;
#endif
    }
  };
  auto stageB = [&](int kt, int p) {
    const long long kofs = (long long)kt * BKK + sko;
    #pragma unroll
    for (int it = 0; it < 4; it++) {
      const int row = srow + it * 64;
      const __bf16* src = Bblk + (long long)row * ldb + kofs;
      __bf16* dst = &sB[p][row][sko];
#if HAVE_ASYNC_LDS
      __builtin_amdgcn_global_load_async_to_lds_b128((g_v4i*)src, (l_v4i*)dst, 0, 0);
#else
      *(bf16x8*)dst = *(const bf16x8*)src;
#endif
    }
  };

  v8f acc[4][4];
  #pragma unroll
  for (int i = 0; i < 4; i++)
    #pragma unroll
    for (int j = 0; j < 4; j++)
      #pragma unroll
      for (int e = 0; e < 8; e++) acc[i][j][e] = 0.0f;

  const int r    = lane & 15;
  const int hi16 = lane >> 4;
  const int kbA  = hi16 * 8;
  const int kbB  = hi16 * 8;

  auto compute = [&](int cur) {
    v16bf afr[4];
    #pragma unroll
    for (int mi = 0; mi < 4; mi++) {
      const bf16x8 lo = *(const bf16x8*)&sA[cur][mw + mi * 16 + r][kbA];
      const bf16x8 hi = *(const bf16x8*)&sA[cur][mw + mi * 16 + r][kbA + 16];
      afr[mi] = __builtin_shufflevector(lo, hi, 0,1,2,3,4,5,6,7,8,9,10,11,12,13,14,15);
    }
    #pragma unroll
    for (int ni = 0; ni < 4; ni++) {
      const int ncol = nw + ni * 16 + r;
      const bf16x8 lo = *(const bf16x8*)&sB[cur][ncol][kbB];
      const bf16x8 hi = *(const bf16x8*)&sB[cur][ncol][kbB + 16];
      const v16bf bfr = __builtin_shufflevector(lo, hi, 0,1,2,3,4,5,6,7,8,9,10,11,12,13,14,15);
      #pragma unroll
      for (int mi = 0; mi < 4; mi++) {
        acc[mi][ni] = __builtin_amdgcn_wmma_f32_16x16x32_bf16(
            false, afr[mi], false, bfr, (short)0, acc[mi][ni], false, false);
      }
    }
  };

  const int nt = K / BKK;
  stageA(0, 0); stageB(0, 0);

  for (int kt = 0; kt < nt - 1; kt++) {           // steady state (branch-free body)
    const int cur = kt & 1;
    stageA(kt + 1, cur ^ 1); stageB(kt + 1, cur ^ 1);
    if (kt + 2 < nt) {  // L2 prefetch two tiles ahead
      __builtin_prefetch(Ablk + (long long)srow * lda + (long long)(kt + 2) * BKK, 0, 1);
      __builtin_prefetch(Bblk + (long long)srow * ldb + (long long)(kt + 2) * BKK, 0, 1);
    }
    WAIT_ASYNC(6);      // retire tile kt (6 asyncs/thread in flight for kt+1)
    __syncthreads();
    compute(cur);
    __syncthreads();
  }
  WAIT_ASYNC(0);
  __syncthreads();
  compute((nt - 1) & 1);

  if (OUTMODE == 0) {
    float* C = (float*)Cg + coff;
    #pragma unroll
    for (int mi = 0; mi < 4; mi++) {
      #pragma unroll
      for (int ni = 0; ni < 4; ni++) {
        #pragma unroll
        for (int rr = 0; rr < 8; rr++) {
          const int row = m0 + mw + mi * 16 + rr + 8 * hi16;
          const int col = n0 + nw + ni * 16 + r;
          float v = acc[mi][ni][rr];
          if (HAS_BIAS)  v += bias[row];
          if (FUSE_GELU) v = gelu_exact(v);
          C[(long long)row * ldc + col] = v;
        }
      }
    }
  } else {
    __bf16* C = (__bf16*)Cg + coff;
    #pragma unroll
    for (int mi = 0; mi < 4; mi++) {
      const int rowb = m0 + mw + mi * 16 + 8 * hi16;
      #pragma unroll
      for (int ni = 0; ni < 4; ni++) {
        const int col = n0 + nw + ni * 16 + r;
        bf16x8 pk;
        #pragma unroll
        for (int rr = 0; rr < 8; rr++) {
          float v = acc[mi][ni][rr];
          if (HAS_BIAS)  v += bias[rowb + rr];
          if (FUSE_GELU) v = gelu_exact(v);
          pk[rr] = f2bf(v);
        }
        *(bf16x8*)&C[(long long)col * ldc + rowb] = pk;
      }
    }
  }
}

// ---------------------------------------------------------------------------
// Conversion / elementwise kernels
// ---------------------------------------------------------------------------
__global__ void tobf16_kernel(const float* __restrict__ x, __bf16* __restrict__ y, long long n4)
{
  const long long i = (long long)blockIdx.x * blockDim.x + threadIdx.x;
  if (i >= n4) return;
  const float4 f4 = ((const float4*)x)[i];
  bf16x4 v; v[0] = f2bf(f4.x); v[1] = f2bf(f4.y); v[2] = f2bf(f4.z); v[3] = f2bf(f4.w);
  ((bf16x4*)y)[i] = v;
}

// f32 [Z][Fdim][Ww] -> bf16 [Z][Ww][Fdim] (tiled transpose + convert)
__global__ __launch_bounds__(256) void t_f32_to_bf16_kernel(
    const float* __restrict__ x, __bf16* __restrict__ y, int Fdim)
{
  __shared__ float tile[32][33];
  const int w0 = blockIdx.x * 32, f0 = blockIdx.y * 32, zz = blockIdx.z;
  const float* xb = x + (long long)zz * Fdim * Ww;
  __bf16* yb = y + (long long)zz * Fdim * Ww;
  for (int i = threadIdx.y; i < 32; i += 8)
    tile[i][threadIdx.x] = xb[(long long)(f0 + i) * Ww + w0 + threadIdx.x];
  __syncthreads();
  for (int i = threadIdx.y; i < 32; i += 8)
    yb[(long long)(w0 + i) * Fdim + f0 + threadIdx.x] = f2bf(tile[threadIdx.x][i]);
}

template<bool APPLY_GELU>
__global__ void conv3x3_kernel(const float* __restrict__ x, const float* __restrict__ wgt,
                               const float* __restrict__ bias, float* __restrict__ y)
{
  const long long total = (long long)Bb * Cc * Ff * Ww;
  long long idx = (long long)blockIdx.x * blockDim.x + threadIdx.x;
  if (idx >= total) return;
  const int w = (int)(idx % Ww); long long t = idx / Ww;
  const int f = (int)(t % Ff); t /= Ff;
  const int co = (int)(t % Cc); const int b = (int)(t / Cc);
  float acc = bias[co];
  #pragma unroll
  for (int ci = 0; ci < Cc; ci++) {
    #pragma unroll
    for (int df = 0; df < 3; df++) {
      const int ff = f + df - 1;
      if (ff < 0 || ff >= Ff) continue;
      #pragma unroll
      for (int dw = 0; dw < 3; dw++) {
        const int ww = w + dw - 1;
        if (ww < 0 || ww >= Ww) continue;
        acc += x[(((long long)b * Cc + ci) * Ff + ff) * Ww + ww] *
               wgt[((co * Cc + ci) * 3 + df) * 3 + dw];
      }
    }
  }
  if (APPLY_GELU) acc = gelu_exact(acc);
  y[idx] = acc;
}

__global__ void conv1x7_kernel(const float* __restrict__ x, const float* __restrict__ wgt,
                               const float* __restrict__ bias, float* __restrict__ y)
{
  const long long total = (long long)Bb * Cc * Ff * Ww;
  long long idx = (long long)blockIdx.x * blockDim.x + threadIdx.x;
  if (idx >= total) return;
  const int w = (int)(idx % Ww); long long t = idx / Ww;
  const int f = (int)(t % Ff); t /= Ff;
  const int co = (int)(t % Cc); const int b = (int)(t / Cc);
  float acc = bias[co];
  #pragma unroll
  for (int ci = 0; ci < Cc; ci++) {
    #pragma unroll
    for (int k = 0; k < 7; k++) {
      const int ww = w + k - 3;
      if (ww < 0 || ww >= Ww) continue;
      acc += x[(((long long)b * Cc + ci) * Ff + f) * Ww + ww] * wgt[(co * Cc + ci) * 7 + k];
    }
  }
  y[idx] = acc;
}

__global__ void conv1x1_kernel(const float* __restrict__ x, const float* __restrict__ wgt,
                               const float* __restrict__ bias, float* __restrict__ y)
{
  const long long total = (long long)Bb * Cc * Ff * Ww;
  long long idx = (long long)blockIdx.x * blockDim.x + threadIdx.x;
  if (idx >= total) return;
  const int w = (int)(idx % Ww); long long t = idx / Ww;
  const int f = (int)(t % Ff); t /= Ff;
  const int co = (int)(t % Cc); const int b = (int)(t / Cc);
  float acc = bias[co];
  #pragma unroll
  for (int ci = 0; ci < Cc; ci++)
    acc += x[(((long long)b * Cc + ci) * Ff + f) * Ww + w] * wgt[co * Cc + ci];
  y[idx] = acc;
}

// read f32 q [bc][f][w], apply RoPE over frames, write bf16 transposed [bc][w][f]
__global__ void rope_t_kernel(const float* __restrict__ q, __bf16* __restrict__ yt)
{
  const long long total = (long long)Bb * Cc * Hh * (Dd / 2) * Ww;
  long long idx = (long long)blockIdx.x * blockDim.x + threadIdx.x;
  if (idx >= total) return;
  const int w = (int)(idx % Ww); long long t = idx / Ww;
  const int j = (int)(t % (Dd / 2)); t /= (Dd / 2);
  const int h = (int)(t % Hh); const int bc = (int)(t / Hh);
  const long long base = ((long long)bc * Ff + h * Dd + 2 * j) * Ww + w;
  const float x0 = q[base], x1 = q[base + Ww];
  const float ang = (float)w * __powf(10000.0f, -2.0f * (float)j / (float)Dd);
  float sn, cs; __sincosf(ang, &sn, &cs);
  bf16x2 o; o[0] = f2bf(x0 * cs - x1 * sn); o[1] = f2bf(x1 * cs + x0 * sn);
  *(bf16x2*)&yt[(long long)bc * Ff * Ww + (long long)w * Ff + h * Dd + 2 * j] = o;
}

__global__ __launch_bounds__(256) void softmax_bf16_kernel(
    const float* __restrict__ s, __bf16* __restrict__ pout, float scale)
{
  const float* p = s + (long long)blockIdx.x * Ww;
  __bf16* po = pout + (long long)blockIdx.x * Ww;
  __shared__ float red[256];
  const int tid = threadIdx.x;
  float v[4];
  float mx = -3.4e38f;
  #pragma unroll
  for (int i = 0; i < 4; i++) { v[i] = p[tid + i * 256] * scale; mx = fmaxf(mx, v[i]); }
  red[tid] = mx; __syncthreads();
  for (int o = 128; o > 0; o >>= 1) { if (tid < o) red[tid] = fmaxf(red[tid], red[tid + o]); __syncthreads(); }
  mx = red[0]; __syncthreads();
  float sum = 0.0f;
  #pragma unroll
  for (int i = 0; i < 4; i++) { v[i] = __expf(v[i] - mx); sum += v[i]; }
  red[tid] = sum; __syncthreads();
  for (int o = 128; o > 0; o >>= 1) { if (tid < o) red[tid] += red[tid + o]; __syncthreads(); }
  const float inv = 1.0f / red[0];
  #pragma unroll
  for (int i = 0; i < 4; i++) po[tid + i * 256] = f2bf(v[i] * inv);
}

__global__ __launch_bounds__(256) void ln_residual_kernel(
    float* __restrict__ h, const float* __restrict__ z,
    const float* __restrict__ g, const float* __restrict__ bt)
{
  __shared__ float s1[8][33];
  __shared__ float s2[8][33];
  __shared__ float smu[32], srs[32];
  const int bc = blockIdx.y;
  const int c  = bc % Cc;
  const int w  = blockIdx.x * 32 + threadIdx.x;
  const long long base = (long long)bc * Ff * Ww + w;
  float sum = 0.0f, sq = 0.0f;
  for (int f = threadIdx.y; f < Ff; f += 8) {
    const float v = h[base + (long long)f * Ww] + z[base + (long long)f * Ww];
    sum += v; sq += v * v;
  }
  s1[threadIdx.y][threadIdx.x] = sum;
  s2[threadIdx.y][threadIdx.x] = sq;
  __syncthreads();
  if (threadIdx.y == 0) {
    float a = 0.0f, b2 = 0.0f;
    #pragma unroll
    for (int i = 0; i < 8; i++) { a += s1[i][threadIdx.x]; b2 += s2[i][threadIdx.x]; }
    const float mu  = a / (float)Ff;
    const float var = b2 / (float)Ff - mu * mu;
    smu[threadIdx.x] = mu;
    srs[threadIdx.x] = rsqrtf(var + 1e-5f);
  }
  __syncthreads();
  const float mu = smu[threadIdx.x], rs = srs[threadIdx.x];
  for (int f = threadIdx.y; f < Ff; f += 8) {
    const long long idx = base + (long long)f * Ww;
    const float v = h[idx] + z[idx];
    h[idx] = (v - mu) * rs * g[c * Ff + f] + bt[c * Ff + f];
  }
}

__global__ void outmix_kernel(const float* __restrict__ h, const float* __restrict__ ow,
                              float* __restrict__ out)
{
  const long long total = (long long)Bb * OCc * Ff * Ww;
  long long idx = (long long)blockIdx.x * blockDim.x + threadIdx.x;
  if (idx >= total) return;
  const int w = (int)(idx % Ww); long long t = idx / Ww;
  const int f = (int)(t % Ff); t /= Ff;
  const int o = (int)(t % OCc); const int b = (int)(t / OCc);
  float acc = 0.0f;
  #pragma unroll
  for (int c = 0; c < Cc; c++)
    acc += h[(((long long)b * Cc + c) * Ff + f) * Ww + w] * ow[o * Cc + c];
  out[idx] = acc;
}

// ---------------------------------------------------------------------------
extern "C" void kernel_launch(void* const* d_in, const int* in_sizes, int n_in,
                              void* d_out, int out_size, void* d_ws, size_t ws_size,
                              hipStream_t stream)
{
  (void)in_sizes; (void)n_in; (void)out_size; (void)ws_size;
  const float* x       = (const float*)d_in[0];
  const float* enc_c1w = (const float*)d_in[1];
  const float* enc_c1b = (const float*)d_in[2];
  const float* enc_c2w = (const float*)d_in[3];
  const float* enc_c2b = (const float*)d_in[4];
  const float* enc_nw  = (const float*)d_in[5];
  const float* enc_nb  = (const float*)d_in[6];
  const float* qw  = (const float*)d_in[7];
  const float* qb  = (const float*)d_in[8];
  const float* qcw = (const float*)d_in[9];
  const float* qcb = (const float*)d_in[10];
  const float* kw  = (const float*)d_in[11];
  const float* kb  = (const float*)d_in[12];
  const float* kcw = (const float*)d_in[13];
  const float* kcb = (const float*)d_in[14];
  const float* vw  = (const float*)d_in[15];
  const float* vb_ = (const float*)d_in[16];
  const float* vcw = (const float*)d_in[17];
  const float* vcb = (const float*)d_in[18];
  const float* ow  = (const float*)d_in[19];
  const float* ob  = (const float*)d_in[20];
  const float* ocw = (const float*)d_in[21];
  const float* ocb = (const float*)d_in[22];
  const float* n1w = (const float*)d_in[23];
  const float* n1b = (const float*)d_in[24];
  const float* f1w = (const float*)d_in[25];
  const float* f1b = (const float*)d_in[26];
  const float* f2w = (const float*)d_in[27];
  const float* f2b = (const float*)d_in[28];
  const float* n2w = (const float*)d_in[29];
  const float* n2b = (const float*)d_in[30];
  const float* out_w = (const float*)d_in[31];

  const long long FW = (long long)Ff * Ww;  // 1M
  float* h  = (float*)d_ws;      // [B,C,F,W]  4M f32
  float* t0 = h  + 4 * FW;       // scratch    4M f32
  float* t1 = t0 + 4 * FW;       // scratch    4M f32
  float* qf = t1 + 4 * FW;       // 4M f32
  float* kf = qf + 4 * FW;       // 4M f32
  float* vf = kf + 4 * FW;       // 4M f32
  float* sc = vf + 4 * FW;       // scores 16M f32
  __bf16* hbt = (__bf16*)(sc + 16 * FW);  // h transposed [bc][w][f]  4M bf16
  __bf16* qt  = hbt + 4 * FW;    // roped q, [bc][w][f]   4M
  __bf16* ktb = qt  + 4 * FW;    // roped k, [bc][w][f]   4M
  __bf16* vbb = ktb + 4 * FW;    // v, [bc][f][w]         4M
  __bf16* abt = vbb + 4 * FW;    // attn out [bc][w][f]   4M
  __bf16* Pb  = abt + 4 * FW;    // softmax probs         16M
  __bf16* fft = Pb  + 16 * FW;   // ffn mid [bc][w][g]    16M
  __bf16* wbs = fft + 16 * FW;   // weight scratch        8M

  (void)hipMemcpyAsync(h, x, sizeof(float) * 4 * FW, hipMemcpyDeviceToDevice, stream);

  const int eg = (int)((4 * FW + 255) / 256);
  const dim3 eb(256);
  const dim3 lnGrid(Ww / 32, Bb * Cc);
  const dim3 lnBlk(32, 8);
  const int ABATCH = Bb * Cc * Hh;   // 16

  auto cvt = [&](const float* src, __bf16* dst, long long n) {
    const long long n4 = n / 4;
    tobf16_kernel<<<(int)((n4 + 255) / 256), 256, 0, stream>>>(src, dst, n4);
  };
  auto cvtT = [&](const float* src, __bf16* dst, int Fdim, int Z) {
    t_f32_to_bf16_kernel<<<dim3(Ww / 32, Fdim / 32, Z), dim3(32, 8), 0, stream>>>(src, dst, Fdim);
  };
  // per-channel linear: weights f32 -> bf16 scratch, then WMMA GEMM
  auto lin = [&](const float* Wf32, const float* bias, const __bf16* Bt, void* Y,
                 int G, int Kd, bool f1mode) {
    cvt(Wf32, wbs, (long long)Cc * G * Kd);
    dim3 grid(Ww / BN, G / BM, Bb * Cc);
    if (f1mode)  // fused GELU, bf16 transposed output [w][g]
      gemm_bf16_wmma<true, true, 1><<<grid, 256, 0, stream>>>(
          wbs, Bt, bias, Y, G, Ww, Kd, Kd, Kd, G,
          Cc, 0, (long long)G * Kd,
          1, (long long)Ww * Kd, 0,
          1, (long long)Ww * G, 0,
          G);
    else         // f32 normal output [g][w]
      gemm_bf16_wmma<false, true, 0><<<grid, 256, 0, stream>>>(
          wbs, Bt, bias, Y, G, Ww, Kd, Kd, Kd, Ww,
          Cc, 0, (long long)G * Kd,
          1, (long long)Ww * Kd, 0,
          1, (long long)G * Ww, 0,
          G);
  };

  // ---- FrameEncoder: 3 ResBlocks ----
  for (int i = 0; i < NBb; i++) {
    conv3x3_kernel<true ><<<eg, eb, 0, stream>>>(h,  enc_c1w + i * Cc * Cc * 9, enc_c1b + i * Cc, t0);
    conv3x3_kernel<false><<<eg, eb, 0, stream>>>(t0, enc_c2w + i * Cc * Cc * 9, enc_c2b + i * Cc, t1);
    ln_residual_kernel<<<lnGrid, lnBlk, 0, stream>>>(h, t1, enc_nw + i * Cc * Ff, enc_nb + i * Cc * Ff);
  }

  // ---- 15 transformer layers ----
  for (int l = 0; l < Ll; l++) {
    const long long wOff = (long long)l * Cc * Ff * Ff;
    cvtT(h, hbt, Ff, Bb * Cc);   // h -> bf16 [w][f] for q/k/v linears

    // Q
    lin(qw + wOff, qb + l * Cc * Ff, hbt, t0, Ff, Ff, false);
    conv1x7_kernel<<<eg, eb, 0, stream>>>(t0, qcw + l * Cc * Cc * 7, qcb + l * Cc, qf);
    rope_t_kernel<<<(int)((2 * FW + 255) / 256), 256, 0, stream>>>(qf, qt);
    // K
    lin(kw + wOff, kb + l * Cc * Ff, hbt, t0, Ff, Ff, false);
    conv1x7_kernel<<<eg, eb, 0, stream>>>(t0, kcw + l * Cc * Cc * 7, kcb + l * Cc, kf);
    rope_t_kernel<<<(int)((2 * FW + 255) / 256), 256, 0, stream>>>(kf, ktb);
    // V
    lin(vw + wOff, vb_ + l * Cc * Ff, hbt, t0, Ff, Ff, false);
    conv1x7_kernel<<<eg, eb, 0, stream>>>(t0, vcw + l * Cc * Cc * 7, vcb + l * Cc, vf);
    cvt(vf, vbb, 4 * FW);

    // scores[z] (WxW f32) = q_t[z] (WxD, lda=F) @ k_t[z]^T (NxK, ldb=F)
    {
      dim3 grid(Ww / BN, Ww / BM, ABATCH);
      gemm_bf16_wmma<false, false, 0><<<grid, 256, 0, stream>>>(
          qt, ktb, nullptr, sc, Ww, Ww, Dd, Ff, Ff, Ww,
          Hh, (long long)Ww * Ff, Dd,
          Hh, (long long)Ww * Ff, Dd,
          1, (long long)Ww * Ww, 0,
          0);
    }
    softmax_bf16_kernel<<<ABATCH * Ww, 256, 0, stream>>>(sc, Pb, 0.03125f);  // 1/sqrt(F)
    // attn out [w][f] bf16: M=D rows of V, N=W frames, K=W; A=V (DxW), B=P (NxK); OUT transposed
    {
      dim3 grid(Ww / BN, Dd / BM, ABATCH);
      gemm_bf16_wmma<false, false, 1><<<grid, 256, 0, stream>>>(
          vbb, Pb, nullptr, abt, Dd, Ww, Ww, Ww, Ww, Ff,
          Hh, (long long)Ff * Ww, (long long)Dd * Ww,
          1, (long long)Ww * Ww, 0,
          Hh, (long long)Ww * Ff, Dd,
          0);
    }

    // output projection (B = attn-out transposed bf16) + 1x1 conv, then LN1
    lin(ow + wOff, ob + l * Cc * Ff, abt, t0, Ff, Ff, false);
    conv1x1_kernel<<<eg, eb, 0, stream>>>(t0, ocw + l * Cc * Cc, ocb + l * Cc, t1);
    ln_residual_kernel<<<lnGrid, lnBlk, 0, stream>>>(h, t1, n1w + l * Cc * Ff, n1b + l * Cc * Ff);

    // FFN
    cvtT(h, hbt, Ff, Bb * Cc);
    lin(f1w + (long long)l * Cc * Ee * Ff * Ff, f1b + (long long)l * Cc * Ee * Ff,
        hbt, fft, Ee * Ff, Ff, true);                 // GELU + bf16 [w][g]
    lin(f2w + (long long)l * Cc * Ff * Ee * Ff, f2b + (long long)l * Cc * Ff,
        fft, t1, Ff, Ee * Ff, false);
    ln_residual_kernel<<<lnGrid, lnBlk, 0, stream>>>(h, t1, n2w + l * Cc * Ff, n2b + l * Cc * Ff);
  }

  // ---- output channel mix ----
  outmix_kernel<<<(int)(((long long)Bb * OCc * FW + 255) / 256), 256, 0, stream>>>(
      h, out_w, (float*)d_out);
}
